// MSDeformableAttentionV2_33200097198144
// MI455X (gfx1250) — matrix operand order repaired and checked by reading, not compiled, not run on hardware
//
#include <hip/hip_runtime.h>

typedef __attribute__((ext_vector_type(16))) _Float16 v16h;
typedef __attribute__((ext_vector_type(8)))  float    v8f;
typedef unsigned int u32x4 __attribute__((ext_vector_type(4)));
typedef int          i32x8 __attribute__((ext_vector_type(8)));
typedef int          i32x4 __attribute__((ext_vector_type(4)));

#define BS    8
#define LQ    2048
#define NH    8
#define HD    32
#define ED    256
#define SUMP  12
#define LTOT  8400           // 80*80 + 40*40 + 20*20
#define NOFF  192            // NH * SUMP * 2
#define NATT  96             // NH * SUMP
#define NTOT  288            // NOFF + NATT
#define MTOT  (BS * LQ)      // 16384 queries

// --------------------------------------------------------------------------
// Kernel 0: transpose value (b,h,c,L) -> (b,h,L,c).
// A 32(L) x 32(c) tile is DMA'd into LDS by the Tensor Data Mover
// (tensor_load_to_lds, D# per ISA ch.8): tile rows padded +1 dword so the
// transposed LDS reads are bank-conflict-free; tensor_dim0 = remaining L so
// the 8400 % 32 tail is zero-filled by TDM OOB handling.
// --------------------------------------------------------------------------
__global__ void __launch_bounds__(256)
vtrans_kernel(const float* __restrict__ v, float* __restrict__ vT) {
  __shared__ float tile[32][33];        // [channel][l_local], 33 = TDM pad
  const int bh = blockIdx.y;            // 0..63  (b*NH + h)
  const int l0 = blockIdx.x * 32;       // position along L
  const int tx = threadIdx.x & 31;
  const int ty = threadIdx.x >> 5;      // 0..7
  const size_t src_elem = (size_t)bh * HD * LTOT + l0;  // tile origin (elements)

#if __has_builtin(__builtin_amdgcn_tensor_load_to_lds)
  if (threadIdx.x < 32) {               // one wave issues the TDM op (EXEC ignored)
    const unsigned long long gaddr = (unsigned long long)(const void*)(v + src_elem);
    const unsigned lds_addr = (unsigned)(size_t)(&tile[0][0]);
    const unsigned dim0 = (unsigned)(LTOT - l0);         // remaining L -> OOB zero-fill

    u32x4 g0; i32x8 g1;
    i32x4 g2 = {0, 0, 0, 0}, g3 = {0, 0, 0, 0};         // 2D tensor: groups 2/3 unused
    // group0: count=1 | lds_addr | global_addr(57) | type=2 ("image")
    g0[0] = 1u;
    g0[1] = lds_addr;
    g0[2] = (unsigned)(gaddr & 0xFFFFFFFFu);
    g0[3] = (unsigned)((gaddr >> 32) & 0x01FFFFFFu) | 0x80000000u;
    // group1: data_size=4B(2<<16) | pad_enable(1<<20) | pad_interval=32dw(4<<22) | pad=1dw(0<<25)
    g1[0] = (int)((2u << 16) | (1u << 20) | (4u << 22));
    g1[1] = (int)((dim0 & 0xFFFFu) << 16);               // tensor_dim0[15:0]
    g1[2] = (int)(((dim0 >> 16) & 0xFFFFu) | ((unsigned)HD << 16)); // dim0 hi | tensor_dim1=32
    g1[3] = (int)(32u << 16);                            // tile_dim0 = 32 (L)
    g1[4] = 32;                                          // tile_dim1 = 32 (c)
    g1[5] = LTOT;                                        // tensor_dim0_stride[31:0]
    g1[6] = 0;                                           // stride0 hi | stride1 lo
    g1[7] = 0;
#if defined(__clang_major__) && __clang_major__ >= 23
    i32x8 g4 = {0, 0, 0, 0, 0, 0, 0, 0};
    __builtin_amdgcn_tensor_load_to_lds(g0, g1, g2, g3, g4, 0);
#else
    __builtin_amdgcn_tensor_load_to_lds(g0, g1, g2, g3, 0);
#endif
    __builtin_amdgcn_s_wait_tensorcnt((short)0);
  }
  __syncthreads();
#else
  // Fallback: manual strided load into LDS
  const size_t src_base = (size_t)bh * HD * LTOT;
  #pragma unroll
  for (int r = 0; r < 32; r += 8) {
    const int c = ty + r;
    const int l = l0 + tx;
    tile[c][tx] = (l < LTOT) ? v[src_base + (size_t)c * LTOT + l] : 0.0f;
  }
  __syncthreads();
#endif

  const size_t dst_base = (size_t)bh * LTOT * HD;
  #pragma unroll
  for (int r = 0; r < 32; r += 8) {
    const int l = l0 + ty + r;
    if (l < LTOT) vT[dst_base + (size_t)l * HD + tx] = tile[tx][ty + r];
  }
}

// --------------------------------------------------------------------------
// Kernel 1: fused projection GEMM  P[16384 x 288] = Q[16384 x 256] x
// [W_off | W_attn] + [b_off | b_attn] via v_wmma_f32_16x16x32_f16.
// Block = 192 threads (6 waves), M-block = 64 rows staged once in LDS (f16).
// grid.y = 3 so 6 waves x 3 = exactly 18 N tiles of 16. Each wave keeps 4
// accumulators (4 M-subtiles) and reuses each B fragment 4x; K fully
// unrolled -> 32 WMMA per wave.
// --------------------------------------------------------------------------
__global__ void __launch_bounds__(192)
proj_kernel(const float* __restrict__ q,
            const float* __restrict__ Woff, const float* __restrict__ boff,
            const float* __restrict__ Watt, const float* __restrict__ batt,
            float* __restrict__ proj) {
  __shared__ _Float16 qs[64][ED];       // 32 KB query tile in f16
  const int lane = threadIdx.x & 31;
  const int wave = threadIdx.x >> 5;    // 0..5
  const int m0   = blockIdx.x * 64;

  // Stage 64x256 f32 query tile -> f16 LDS (all 192 threads cooperate).
  for (int i = threadIdx.x; i < 64 * ED; i += 192) {
    const int r = i >> 8, c = i & 255;
    qs[r][c] = (_Float16)q[(size_t)(m0 + r) * ED + c];
  }
  __syncthreads();

  const int ntile = blockIdx.y * 6 + wave;   // 0..17, exact cover
  const int n0 = ntile * 16;
  const float* W; int ldW, col;
  if (n0 < NOFF) { W = Woff; ldW = NOFF; col = n0;        }  // W_off (256x192)
  else           { W = Watt; ldW = NATT; col = n0 - NOFF; }  // W_attn (256x96)

  v8f acc0 = {}, acc1 = {}, acc2 = {}, acc3 = {};
  const int arow  = lane & 15;               // A: M = lane%16
  const int khalf = (lane >> 4) * 8;         // A: lanes 16-31 hold K+8 group
  #pragma unroll
  for (int k0 = 0; k0 < ED; k0 += 32) {
    // B fragment (32x16 f16): N = lane%16, elem i<8 -> K=khalf+i, i>=8 -> K=khalf+8+i
    v16h b;
    const float* wrow = W + (size_t)k0 * ldW + col + (lane & 15);
    #pragma unroll
    for (int i = 0; i < 16; ++i) b[i] = (_Float16)wrow[(size_t)(khalf + ((i < 8) ? i : (i + 8))) * ldW];

    #pragma unroll
    for (int mt = 0; mt < 4; ++mt) {
      // A fragment (16x32 f16): elem i<8 -> K=khalf+i, i>=8 -> K=khalf+8+i
      v16h a;
      #pragma unroll
      for (int i = 0; i < 16; ++i) {
        const int kk = khalf + ((i < 8) ? i : (i + 8));
        a[i] = qs[mt * 16 + arow][k0 + kk];
      }
      v8f& acc = (mt == 0) ? acc0 : (mt == 1) ? acc1 : (mt == 2) ? acc2 : acc3;
      acc = __builtin_amdgcn_wmma_f32_16x16x32_f16(
          /*neg_a=*/false, a, /*neg_b=*/false, b,
          /*c_mod=*/(short)0, acc, /*reuse_a=*/false, /*reuse_b=*/false);
    }
  }

  // C/D layout: VGPR j -> M = j + 8*(lane/16), N = lane%16. Fuse bias on store.
  const int n  = lane & 15;
  const int gn = n0 + n;
  const float bias = (gn < NOFF) ? boff[gn] : batt[gn - NOFF];
  const int mhi = (lane >> 4) * 8;
  #pragma unroll
  for (int mt = 0; mt < 4; ++mt) {
    const v8f acc = (mt == 0) ? acc0 : (mt == 1) ? acc1 : (mt == 2) ? acc2 : acc3;
    #pragma unroll
    for (int j = 0; j < 8; ++j) {
      const int m = m0 + mt * 16 + mhi + j;
      proj[(size_t)m * NTOT + gn] = acc[j] + bias;
    }
  }
}

// --------------------------------------------------------------------------
// Kernel 2: softmax over 12 logits + bilinear sampling + weighted sum.
// Block = one query (256 threads = 8 waves); wave = head, lane = channel.
// Sampling coordinates are wave-uniform -> divergence-free; channel gathers
// are contiguous 128B lines in transposed value (L2-resident, 68.8MB<192MB).
// --------------------------------------------------------------------------
__global__ void __launch_bounds__(256)
sample_kernel(const float* __restrict__ proj, const float* __restrict__ rp,
              const float* __restrict__ vT, float* __restrict__ out) {
  const int qg   = blockIdx.x;          // b*LQ + q
  const int h    = threadIdx.x >> 5;    // head
  const int lane = threadIdx.x & 31;    // channel
  const int b    = qg >> 11;            // qg / LQ

  const float* pq = proj + (size_t)qg * NTOT;
  const float rx = rp[qg * 4 + 0], ry = rp[qg * 4 + 1];
  const float rw = rp[qg * 4 + 2], rh = rp[qg * 4 + 3];

  // softmax over the 12 attention logits of this head
  float lg[SUMP];
  float mx = -1e30f;
  #pragma unroll
  for (int p = 0; p < SUMP; ++p) {
    lg[p] = pq[NOFF + h * SUMP + p];
    mx = fmaxf(mx, lg[p]);
  }
  float den = 0.0f;
  #pragma unroll
  for (int p = 0; p < SUMP; ++p) { lg[p] = __expf(lg[p] - mx); den += lg[p]; }
  const float inv = 1.0f / den;

  const int lvlH[3] = {80, 40, 20};
  const int lvlW[3] = {80, 40, 20};
  const int vst[3]  = {0, 6400, 8000};
  const size_t vbase = (size_t)(b * NH + h) * LTOT;

  float acc = 0.0f;
  #pragma unroll
  for (int p = 0; p < SUMP; ++p) {
    const int lvl = p >> 2;                  // 4 points per level
    const int Hh = lvlH[lvl], Ww = lvlW[lvl];
    const float ox = pq[h * SUMP * 2 + 2 * p + 0];
    const float oy = pq[h * SUMP * 2 + 2 * p + 1];
    // NUM_POINTS_SCALE == 0.25 for every point; grid->pixel: x = loc*W - 0.5
    const float locx = rx + ox * 0.25f * rw * 0.5f;
    const float locy = ry + oy * 0.25f * rh * 0.5f;
    const float x = locx * (float)Ww - 0.5f;
    const float y = locy * (float)Hh - 0.5f;
    const float x0f = floorf(x), y0f = floorf(y);
    const float wx1 = x - x0f, wy1 = y - y0f;
    const float wx0 = 1.0f - wx1, wy0 = 1.0f - wy1;
    const int x0 = (int)x0f, y0 = (int)y0f;
    const int x1 = x0 + 1, y1 = y0 + 1;

    float s = 0.0f;
    const size_t lbase = vbase + (size_t)vst[lvl];
    // wave-uniform bounds checks (zero-pad out-of-range, matches reference)
    if (y0 >= 0 && y0 < Hh) {
      const size_t rbase = (lbase + (size_t)y0 * Ww) * HD + lane;
      if (x0 >= 0 && x0 < Ww) s += (wx0 * wy0) * vT[rbase + (size_t)x0 * HD];
      if (x1 >= 0 && x1 < Ww) s += (wx1 * wy0) * vT[rbase + (size_t)x1 * HD];
    }
    if (y1 >= 0 && y1 < Hh) {
      const size_t rbase = (lbase + (size_t)y1 * Ww) * HD + lane;
      if (x0 >= 0 && x0 < Ww) s += (wx0 * wy1) * vT[rbase + (size_t)x0 * HD];
      if (x1 >= 0 && x1 < Ww) s += (wx1 * wy1) * vT[rbase + (size_t)x1 * HD];
    }
    acc += lg[p] * s;
  }

  out[(size_t)qg * ED + h * HD + lane] = acc * inv;
}

// --------------------------------------------------------------------------
extern "C" void kernel_launch(void* const* d_in, const int* in_sizes, int n_in,
                              void* d_out, int out_size, void* d_ws, size_t ws_size,
                              hipStream_t stream) {
  const float* query = (const float*)d_in[0];  // (8,2048,256)
  const float* refp  = (const float*)d_in[1];  // (8,2048,1,4)
  const float* value = (const float*)d_in[2];  // (8,8,32,8400)
  const float* Woff  = (const float*)d_in[3];  // (256,192)
  const float* boff  = (const float*)d_in[4];  // (192,)
  const float* Watt  = (const float*)d_in[5];  // (256,96)
  const float* batt  = (const float*)d_in[6];  // (96,)
  float* out = (float*)d_out;                  // (8,2048,256)

  float* vT   = (float*)d_ws;                          // 8*8*8400*32 f32 = 68.8 MB
  float* proj = vT + (size_t)BS * NH * LTOT * HD;      // 16384*288 f32   = 18.9 MB

  // 1) TDM tile transpose of value for coalesced channel gathers
  vtrans_kernel<<<dim3((LTOT + 31) / 32, BS * NH), 256, 0, stream>>>(value, vT);
  // 2) WMMA projection GEMM (offsets + attention logits, bias fused)
  proj_kernel<<<dim3(MTOT / 64, 3), 192, 0, stream>>>(query, Woff, boff, Watt, batt, proj);
  // 3) softmax + bilinear sampling + weighted reduction
  sample_kernel<<<dim3(MTOT), 256, 0, stream>>>(proj, refp, vT, out);
}
